// GhostMLP_42305427866015
// MI455X (gfx1250) — 8-bit matrix operand order repaired and checked by reading, compiled, not run on hardware
//
#include <hip/hip_runtime.h>
#include <math.h>

typedef __attribute__((ext_vector_type(8))) int v8i;

#define EPSQ 1e-5f
#define TPB 256

__global__ void absum_partial(const float* __restrict__ w, float* __restrict__ part, int n_per_block) {
    __shared__ float sm[TPB];
    const float* p = w + (size_t)blockIdx.x * n_per_block;
    float s = 0.f;
    for (int i = threadIdx.x; i < n_per_block; i += TPB) s += fabsf(p[i]);
    sm[threadIdx.x] = s;
    __syncthreads();
    for (int off = TPB / 2; off > 0; off >>= 1) {
        if ((int)threadIdx.x < off) sm[threadIdx.x] += sm[threadIdx.x + off];
        __syncthreads();
    }
    if (threadIdx.x == 0) part[blockIdx.x] = sm[0];
}

__global__ void absum_finalize(const float* __restrict__ part, int nparts, float invN,
                               float* __restrict__ out) {
    __shared__ float sm[TPB];
    float s = 0.f;
    for (int i = threadIdx.x; i < nparts; i += TPB) s += part[i];
    sm[threadIdx.x] = s;
    __syncthreads();
    for (int off = TPB / 2; off > 0; off >>= 1) {
        if ((int)threadIdx.x < off) sm[threadIdx.x] += sm[threadIdx.x + off];
        __syncthreads();
    }
    if (threadIdx.x == 0) *out = fmaxf(sm[0] * invN, EPSQ);
}

__global__ void row_absmax(const float* __restrict__ x, float* __restrict__ amax, int K) {
    __shared__ float sm[TPB];
    const float* p = x + (size_t)blockIdx.x * K;
    float m = 0.f;
    for (int i = threadIdx.x; i < K; i += TPB) m = fmaxf(m, fabsf(p[i]));
    sm[threadIdx.x] = m;
    __syncthreads();
    for (int off = TPB / 2; off > 0; off >>= 1) {
        if ((int)threadIdx.x < off) sm[threadIdx.x] = fmaxf(sm[threadIdx.x], sm[threadIdx.x + off]);
        __syncthreads();
    }
    if (threadIdx.x == 0) amax[blockIdx.x] = fmaxf(sm[0], EPSQ);
}

__global__ void zero_i32(int* __restrict__ p) { p[blockIdx.x * TPB + threadIdx.x] = 0; }

__global__ void fin_hmax(const int* __restrict__ hmax, float* __restrict__ amax) {
    int i = blockIdx.x * TPB + threadIdx.x;
    amax[i] = fmaxf(__int_as_float(hmax[i]), EPSQ);
}

__global__ void quant_weight_bswz(const float* __restrict__ w, const float* __restrict__ sptr,
                                  signed char* __restrict__ dst, int N, int K) {
    int t = blockIdx.x * TPB + threadIdx.x;
    int kq = K >> 2;
    int n  = t / kq;
    int k  = (t - n * kq) << 2;
    float inv_s = 1.f / *sptr;
    const float4 wv = *reinterpret_cast<const float4*>(w + (size_t)n * K + k);
    int q0 = (int)fminf(fmaxf(rintf(wv.x * inv_s), -1.f), 1.f);
    int q1 = (int)fminf(fmaxf(rintf(wv.y * inv_s), -1.f), 1.f);
    int q2 = (int)fminf(fmaxf(rintf(wv.z * inv_s), -1.f), 1.f);
    int q3 = (int)fminf(fmaxf(rintf(wv.w * inv_s), -1.f), 1.f);
    unsigned pack = (unsigned)(q0 & 255) | ((unsigned)(q1 & 255) << 8) |
                    ((unsigned)(q2 & 255) << 16) | ((unsigned)(q3 & 255) << 24);
    size_t tile = (size_t)(n >> 4) * (K >> 6) + (k >> 6);
    int lane = (n & 15) | (((k >> 4) & 1) << 4);
    int v    = (((k >> 5) & 1) << 2) | ((k >> 2) & 3);
    reinterpret_cast<unsigned*>(dst)[tile * 256 + lane * 8 + v] = pack;
}

__global__ void quant_act_aswz(const float* __restrict__ x, const float* __restrict__ amax,
                               signed char* __restrict__ dst, int M, int K) {
    int t = blockIdx.x * TPB + threadIdx.x;
    int kq = K >> 2;
    int m  = t / kq;
    int k  = (t - m * kq) << 2;
    float scale = 127.f / amax[m];
    const float4 xv = *reinterpret_cast<const float4*>(x + (size_t)m * K + k);
    int q0 = (int)fminf(fmaxf(rintf(xv.x * scale), -128.f), 127.f);
    int q1 = (int)fminf(fmaxf(rintf(xv.y * scale), -128.f), 127.f);
    int q2 = (int)fminf(fmaxf(rintf(xv.z * scale), -128.f), 127.f);
    int q3 = (int)fminf(fmaxf(rintf(xv.w * scale), -128.f), 127.f);
    unsigned pack = (unsigned)(q0 & 255) | ((unsigned)(q1 & 255) << 8) |
                    ((unsigned)(q2 & 255) << 16) | ((unsigned)(q3 & 255) << 24);
    size_t tile = (size_t)(m >> 4) * (K >> 6) + (k >> 6);
    int lane = (m & 15) | (((k >> 4) & 1) << 4);
    int v    = (((k >> 5) & 1) << 2) | ((k >> 2) & 3);
    reinterpret_cast<unsigned*>(dst)[tile * 256 + lane * 8 + v] = pack;
}

template <int FUSE_GELU>
__global__ __launch_bounds__(TPB, 1) void gemm_iu8(
        const signed char* __restrict__ A, const signed char* __restrict__ Bm,
        const float* __restrict__ sptr, const float* __restrict__ amaxRow,
        float* __restrict__ outF, int* __restrict__ hmax,
        int M, int N, int K) {
    const int wid  = (blockIdx.x * blockDim.x + threadIdx.x) >> 5;
    const int lane = threadIdx.x & 31;
    const int nregs  = N >> 6;
    const int mreg   = wid / nregs;
    const int nreg   = wid - mreg * nregs;
    const int ktiles = K >> 6;

    const signed char* ap[4];
    const signed char* bp[4];
#pragma unroll
    for (int mi = 0; mi < 4; ++mi)
        ap[mi] = A + ((size_t)(mreg * 4 + mi) * ktiles) * 1024 + lane * 32;
#pragma unroll
    for (int j = 0; j < 4; ++j)
        bp[j] = Bm + ((size_t)(nreg * 4 + j) * ktiles) * 1024 + lane * 32;

    v8i acc[4][4] = {};
    for (int kt = 0; kt < ktiles; ++kt) {
        const size_t o = (size_t)kt * 1024;
        v8i a[4];
#pragma unroll
        for (int mi = 0; mi < 4; ++mi) a[mi] = *reinterpret_cast<const v8i*>(ap[mi] + o);
#pragma unroll
        for (int j = 0; j < 4; ++j) {
            v8i b = *reinterpret_cast<const v8i*>(bp[j] + o);
#pragma unroll
            for (int mi = 0; mi < 4; ++mi)
                acc[mi][j] = __builtin_amdgcn_wmma_i32_16x16x64_iu8(true, a[mi], true, b,
                                                                    acc[mi][j], false, false);
        }
    }

    const float s    = *sptr;
    const int   nb   = (nreg << 6) + (lane & 15);
    const int   half = lane >> 4;
#pragma unroll
    for (int mi = 0; mi < 4; ++mi) {
#pragma unroll
        for (int r = 0; r < 8; ++r) {
            const int   m = ((mreg * 4 + mi) << 4) + r + (half << 3);
            const float f = s * amaxRow[m] * (1.f / 127.f);
            float lm = 0.f;
#pragma unroll
            for (int j = 0; j < 4; ++j) {
                float v = (float)acc[mi][j][r] * f;
                if (FUSE_GELU) {
                    v = 0.5f * v * (1.f + erff(v * 0.70710678118654752f));
                    lm = fmaxf(lm, fabsf(v));
                }
                outF[(size_t)m * N + nb + (j << 4)] = v;
            }
            if (FUSE_GELU) {
                for (int off = 8; off; off >>= 1) lm = fmaxf(lm, __shfl_xor(lm, off, 16));
                if ((lane & 15) == 0) atomicMax(hmax + m, __float_as_int(lm));
            }
        }
    }
}

extern "C" void kernel_launch(void* const* d_in, const int* in_sizes, int n_in,
                              void* d_out, int out_size, void* d_ws, size_t ws_size,
                              hipStream_t stream) {
    const float* x  = (const float*)d_in[0];
    const float* w1 = (const float*)d_in[1];
    const float* w2 = (const float*)d_in[2];
    float* out = (float*)d_out;

    const int D = 2048, H = 8192, T = 4 * 2048;
    const int NW = H * D;

    char* ws = (char*)d_ws;
    float* s1     = (float*)(ws + 0);
    float* s2     = (float*)(ws + 4);
    float* part   = (float*)(ws + 256);
    float* amax_x = (float*)(ws + (64 << 10));
    int*   hmax   = (int*)  (ws + (96 << 10));
    float* amax_h = (float*)(ws + (128 << 10));
    size_t off = (size_t)256 << 10;
    signed char* w1q = (signed char*)(ws + off); off += (size_t)H * D;
    signed char* w2q = (signed char*)(ws + off); off += (size_t)D * H;
    signed char* xq  = (signed char*)(ws + off); off += (size_t)T * D;
    signed char* hq  = (signed char*)(ws + off); off += (size_t)T * H;
    float*       hbf = (float*)(ws + off);

    absum_partial<<<4096, TPB, 0, stream>>>(w1, part, NW / 4096);
    absum_finalize<<<1, TPB, 0, stream>>>(part, 4096, 1.f / (float)NW, s1);
    absum_partial<<<4096, TPB, 0, stream>>>(w2, part, NW / 4096);
    absum_finalize<<<1, TPB, 0, stream>>>(part, 4096, 1.f / (float)NW, s2);

    quant_weight_bswz<<<(H / 4) * (D / TPB), TPB, 0, stream>>>(w1, s1, w1q, H, D);
    quant_weight_bswz<<<(D / 4) * (H / TPB), TPB, 0, stream>>>(w2, s2, w2q, D, H);

    row_absmax<<<T, TPB, 0, stream>>>(x, amax_x, D);
    quant_act_aswz<<<(T / 4) * (D / TPB), TPB, 0, stream>>>(x, amax_x, xq, T, D);

    zero_i32<<<T / TPB, TPB, 0, stream>>>(hmax);
    {
        int waves = (T / 64) * (H / 64);
        gemm_iu8<1><<<waves * 32 / TPB, TPB, 0, stream>>>(xq, w1q, s1, amax_x, hbf, hmax, T, H, D);
    }

    fin_hmax<<<T / TPB, TPB, 0, stream>>>(hmax, amax_h);
    quant_act_aswz<<<(T / 4) * (H / TPB), TPB, 0, stream>>>(hbf, amax_h, hq, T, H);
    {
        int waves = (T / 64) * (D / 64);
        gemm_iu8<0><<<waves * 32 / TPB, TPB, 0, stream>>>(hq, w2q, s2, amax_h, out, nullptr, T, D, H);
    }
}
